// benchmark_28398323761499
// MI455X (gfx1250) — matrix operand order repaired and checked by reading, not compiled, not run on hardware
//
#include <hip/hip_runtime.h>

// ---------------------------------------------------------------------------
// CDNA5 (gfx1250) types + WMMA helpers
// ---------------------------------------------------------------------------
typedef __attribute__((ext_vector_type(16))) __bf16 v16bf;
typedef __attribute__((ext_vector_type(8)))  __bf16 bf16x8;
typedef __attribute__((ext_vector_type(8)))  float  v8f;

#define V8F_ZERO (v8f){0.f,0.f,0.f,0.f,0.f,0.f,0.f,0.f}

__device__ __forceinline__ v8f wmma_bf16(v16bf a, v16bf b, v8f c) {
  // D = A(16x32 bf16) * B(32x16 bf16) + C(16x16 f32)
  return __builtin_amdgcn_wmma_f32_16x16x32_bf16(false, a, false, b, (short)0, c,
                                                 false, false);
}

__device__ __forceinline__ v16bf ld16(const __bf16* p) {  // 16 contiguous (32B)
  return *(const v16bf*)p;
}
__device__ __forceinline__ v16bf ld8x2(const __bf16* p0, const __bf16* p1) {
  bf16x8 a = *(const bf16x8*)p0;
  bf16x8 b = *(const bf16x8*)p1;
  v16bf r;
#pragma unroll
  for (int i = 0; i < 8; ++i) { r[i] = a[i]; r[i + 8] = b[i]; }
  return r;
}

// ---------------------------------------------------------------------------
// fp32 -> bf16 convert
// ---------------------------------------------------------------------------
__global__ void k_f2b(const float* __restrict__ x, __bf16* __restrict__ y, int n) {
  int i = blockIdx.x * blockDim.x + threadIdx.x;
  if (i < n) y[i] = (__bf16)x[i];
}

// ---------------------------------------------------------------------------
// WMMA GEMM:  C[M,N] = A[M,K] @ W[N,K]^T + bias[N]
// block = 128 threads (4 waves); block tile 64(M)x64(N); wave tile 16x64.
// A fragment layout (16-bit A 16x32): lane<16 -> row=lane, K k..k+15
//                                     lane>=16 -> row=lane-16, K k+16..k+31
// B fragment layout (16-bit B 32x16): lane holds col=lane&15, K 16 contiguous
//                                     (lanes<16 -> K k..k+15, lanes>=16 -> k+16..k+31)
// ---------------------------------------------------------------------------
__global__ void k_gemm(const __bf16* __restrict__ A, const __bf16* __restrict__ W,
                       const float* __restrict__ bias, float* __restrict__ C,
                       int M, int N, int K) {
  int wave = threadIdx.x >> 5;
  int lane = threadIdx.x & 31;
  int l15  = lane & 15;
  int hh   = lane >> 4;
  int n0 = blockIdx.x * 64;
  int m0 = blockIdx.y * 64 + wave * 16;

  v8f acc[4];
#pragma unroll
  for (int t = 0; t < 4; ++t) acc[t] = V8F_ZERO;

  const __bf16* arow = A + (size_t)(m0 + l15) * K + hh * 16;
  for (int k = 0; k < K; k += 32) {
    v16bf af = ld16(arow + k);
#pragma unroll
    for (int t = 0; t < 4; ++t) {
      const __bf16* bp = W + (size_t)(n0 + t * 16 + l15) * K + k + hh * 16;
      acc[t] = wmma_bf16(af, ld16(bp), acc[t]);
    }
  }
  // C/D layout: lane holds col = lane&15; VGPR v -> row v + 8*(lane>=16)
#pragma unroll
  for (int t = 0; t < 4; ++t) {
    int c = n0 + t * 16 + l15;
    float bv = bias ? bias[c] : 0.f;
#pragma unroll
    for (int v = 0; v < 8; ++v) {
      int r = m0 + v + hh * 8;
      C[(size_t)r * N + c] = acc[t][v] + bv;
    }
  }
}

// ---------------------------------------------------------------------------
// Row LayerNorm (wave per row). Optional affine, leaky-relu, f32 + bf16 outputs
// with independent row strides / column offsets (used to build `comb` in-place).
// ---------------------------------------------------------------------------
__global__ void k_ln(const float* __restrict__ X, int W,
                     const float* __restrict__ g, const float* __restrict__ b,
                     float* __restrict__ Yf, int fs, int fc,
                     __bf16* __restrict__ Yb, int bs, int bc,
                     int leaky, int rows) {
  int r = (blockIdx.x * blockDim.x + threadIdx.x) >> 5;
  int lane = threadIdx.x & 31;
  if (r >= rows) return;
  const float* x = X + (size_t)r * W;
  float s = 0.f;
  for (int j = lane; j < W; j += 32) s += x[j];
#pragma unroll
  for (int o = 16; o >= 1; o >>= 1) s += __shfl_xor(s, o, 32);
  float mean = s / (float)W;
  float v = 0.f;
  for (int j = lane; j < W; j += 32) { float d = x[j] - mean; v += d * d; }
#pragma unroll
  for (int o = 16; o >= 1; o >>= 1) v += __shfl_xor(v, o, 32);
  float inv = rsqrtf(v / (float)W + 1e-5f);
  for (int j = lane; j < W; j += 32) {
    float y = (x[j] - mean) * inv;
    if (g) y = y * g[j] + b[j];
    if (leaky && y < 0.f) y *= 0.01f;
    if (Yf) Yf[(size_t)r * fs + fc + j] = y;
    if (Yb) Yb[(size_t)r * bs + bc + j] = (__bf16)y;
  }
}

// Per-head LN (width 32, no affine): wave per (token,head)
__global__ void k_headln(const float* __restrict__ X, float* __restrict__ Y, int rows) {
  int r = (blockIdx.x * blockDim.x + threadIdx.x) >> 5;
  int lane = threadIdx.x & 31;
  if (r >= rows) return;
  float x = X[(size_t)r * 32 + lane];
  float s = x;
#pragma unroll
  for (int o = 16; o >= 1; o >>= 1) s += __shfl_xor(s, o, 32);
  float mean = s * (1.f / 32.f);
  float d = x - mean;
  float v = d * d;
#pragma unroll
  for (int o = 16; o >= 1; o >>= 1) v += __shfl_xor(v, o, 32);
  Y[(size_t)r * 32 + lane] = d * rsqrtf(v * (1.f / 32.f) + 1e-5f);
}

// V transpose+convert: Vt[d][token] = bf16(Vf[token][d]), d = h*32+hd
__global__ void k_vt(const float* __restrict__ Vf, __bf16* __restrict__ Vt) {
  int i = blockIdx.x * blockDim.x + threadIdx.x;
  if (i >= 4096 * 256) return;
  int tok = i >> 8, d = i & 255;
  Vt[(size_t)d * 4096 + tok] = (__bf16)Vf[i];
}

// ---------------------------------------------------------------------------
// KNN: 24 nearest (incl. self) per row; one wave per row, distances in LDS.
// ---------------------------------------------------------------------------
__global__ void k_knn(const float* __restrict__ pos, int* __restrict__ idx) {
  __shared__ float srow[2048];
  int i = blockIdx.x, lane = threadIdx.x;
  float px = pos[i * 3 + 0], py = pos[i * 3 + 1], pz = pos[i * 3 + 2];
  for (int j = lane; j < 2048; j += 32) {
    float dx = pos[j * 3 + 0] - px, dy = pos[j * 3 + 1] - py, dz = pos[j * 3 + 2] - pz;
    srow[j] = dx * dx + dy * dy + dz * dz;
  }
  __syncthreads();
  for (int t = 0; t < 24; ++t) {
    float best = 3.4e38f; int bi = 1 << 30;
    for (int j = lane; j < 2048; j += 32) {
      float v = srow[j];
      if (v < best) { best = v; bi = j; }
    }
#pragma unroll
    for (int o = 16; o >= 1; o >>= 1) {
      float ov = __shfl_xor(best, o, 32);
      int   oi = __shfl_xor(bi, o, 32);
      if (ov < best || (ov == best && oi < bi)) { best = ov; bi = oi; }
    }
    if (lane == 0) { idx[i * 24 + t] = bi; srow[bi] = 3.4e38f; }
    __syncthreads();
  }
}

// ---------------------------------------------------------------------------
// Dense flash attention over the always-unmasked half (shared by all scales).
// Wave = (side, head, 16-query tile). 64 blocks of 32 keys.
// Emits unnormalized O[16x32], row max m, row sumexp l.
// ---------------------------------------------------------------------------
__global__ void k_flash(const __bf16* __restrict__ Qb,  // [2*2048, 256]
                        const __bf16* __restrict__ Kb,  // [4096, 256]
                        const __bf16* __restrict__ Vt,  // [256, 4096]
                        float* __restrict__ Mws, float* __restrict__ Lws,
                        float* __restrict__ Ows) {       // [2*8*2048(*32)]
  __shared__ __attribute__((aligned(32))) __bf16 plds[4 * 16 * 32];
  int wave = threadIdx.x >> 5;
  int lane = threadIdx.x & 31;
  int g = blockIdx.x * 4 + wave;
  int side = g >> 10;
  int rem  = g & 1023;
  int h  = rem >> 7;
  int qt = rem & 127;
  int l15 = lane & 15;
  int hh  = lane >> 4;
  int dbase = side ? 0 : 2048;                       // dense (unmasked) key half
  const float scale = 0.17677669529663687f;          // 1/sqrt(32)

  const __bf16* qp = Qb + (size_t)(side * 2048 + qt * 16 + l15) * 256 + h * 32 + hh * 16;
  v16bf qf = ld16(qp);

  v8f acc0 = V8F_ZERO, acc1 = V8F_ZERO;
  float m[8], l[8];
#pragma unroll
  for (int v = 0; v < 8; ++v) { m[v] = -3.0e38f; l[v] = 0.f; }
  __bf16* pbase = plds + wave * 512;

#pragma unroll 2
  for (int kb = 0; kb < 64; ++kb) {
    int j0 = kb * 32;
    const __bf16* kp0 = Kb + (size_t)(dbase + j0 + l15) * 256 + h * 32 + hh * 16;
    v8f s0 = wmma_bf16(qf, ld16(kp0), V8F_ZERO);
    v8f s1 = wmma_bf16(qf, ld16(kp0 + 16 * 256), V8F_ZERO);

    float p0[8], p1[8];
#pragma unroll
    for (int v = 0; v < 8; ++v) {
      float a = s0[v] * scale, b = s1[v] * scale;
      float mx = fmaxf(a, b);
#pragma unroll
      for (int o = 8; o >= 1; o >>= 1) mx = fmaxf(mx, __shfl_xor(mx, o, 16));
      float mn = fmaxf(m[v], mx);
      float f  = __expf(m[v] - mn);
      float e0 = __expf(a - mn), e1 = __expf(b - mn);
      float rs = e0 + e1;
#pragma unroll
      for (int o = 8; o >= 1; o >>= 1) rs += __shfl_xor(rs, o, 16);
      l[v] = l[v] * f + rs;
      m[v] = mn;
      acc0[v] *= f; acc1[v] *= f;
      p0[v] = e0; p1[v] = e1;
    }
    // C-layout -> LDS (row-major 16x32 bf16) -> reload as A fragment
#pragma unroll
    for (int v = 0; v < 8; ++v) {
      int r = v + hh * 8;
      pbase[r * 32 + l15]      = (__bf16)p0[v];
      pbase[r * 32 + l15 + 16] = (__bf16)p1[v];
    }
    asm volatile("s_wait_dscnt 0" ::: "memory");     // wave-local LDS RAW
    __builtin_amdgcn_wave_barrier();
    const __bf16* pr = pbase + l15 * 32 + hh * 16;
    v16bf pf = ld16(pr);

    const __bf16* vp0 = Vt + (size_t)(h * 32 + l15) * 4096 + dbase + j0 + hh * 16;
    const __bf16* vp1 = vp0 + 16 * 4096;
    acc0 = wmma_bf16(pf, ld16(vp0), acc0);
    acc1 = wmma_bf16(pf, ld16(vp1), acc1);
    __builtin_amdgcn_wave_barrier();
  }

  size_t base = (size_t)(side * 8 + h) * 2048 + qt * 16;
#pragma unroll
  for (int v = 0; v < 8; ++v) {
    int r = v + hh * 8;
    Ows[(base + r) * 32 + l15]      = acc0[v];
    Ows[(base + r) * 32 + l15 + 16] = acc1[v];
  }
  if (l15 == 0) {
#pragma unroll
    for (int v = 0; v < 8; ++v) {
      Mws[base + v + hh * 8] = m[v];
      Lws[base + v + hh * 8] = l[v];
    }
  }
}

// ---------------------------------------------------------------------------
// KNN corrections + 3-scale softmax mix. Block per (side, token); wave = head.
// ---------------------------------------------------------------------------
__global__ void k_mix(const float* __restrict__ Qf,   // [2*2048,256]
                      const float* __restrict__ Kf,   // [4096,256]
                      const float* __restrict__ Vf,   // [4096,256]
                      const int* __restrict__ knn,    // [2048,24] (sorted by dist)
                      const float* __restrict__ Mws, const float* __restrict__ Lws,
                      const float* __restrict__ Ows,
                      const float* __restrict__ swg, const float* __restrict__ sws,
                      float* __restrict__ mix) {      // [2*2048,256]
  int blk = blockIdx.x;
  int side = blk >> 11;
  int i = blk & 2047;
  int h = threadIdx.x >> 5;
  int lane = threadIdx.x & 31;
  int kbase = side ? 2048 : 0;                        // KNN keys = same-modality half
  const float inv = 0.17677669529663687f;

  float q = Qf[(size_t)(side * 2048 + i) * 256 + h * 32 + lane];
  float s[24];
#pragma unroll
  for (int t = 0; t < 24; ++t) {
    int j = knn[i * 24 + t];
    float d = q * Kf[(size_t)(kbase + j) * 256 + h * 32 + lane];
#pragma unroll
    for (int o = 16; o >= 1; o >>= 1) d += __shfl_xor(d, o, 32);
    s[t] = d * inv;
  }
  size_t sb = (size_t)(side * 8 + h) * 2048 + i;
  float md = Mws[sb], ld = Lws[sb];
  float o0 = Ows[sb * 32 + lane];
  float mt = md;
#pragma unroll
  for (int t = 0; t < 24; ++t) mt = fmaxf(mt, s[t]);
  float f = __expf(md - mt);
  float lsum = ld * f;
  float acc  = o0 * f;

  const float* sw = side ? sws : swg;
  float e0 = __expf(sw[0]), e1 = __expf(sw[1]), e2 = __expf(sw[2]);
  float es = e0 + e1 + e2;
  float w0 = e0 / es, w1 = e1 / es, w2 = e2 / es;

  float out = 0.f;
#pragma unroll
  for (int t = 0; t < 24; ++t) {
    int j = knn[i * 24 + t];
    float e = __expf(s[t] - mt);
    lsum += e;
    acc  += e * Vf[(size_t)(kbase + j) * 256 + h * 32 + lane];
    if (t == 7)  out += w0 * (acc / lsum);
    if (t == 15) out += w1 * (acc / lsum);
    if (t == 23) out += w2 * (acc / lsum);
  }
  mix[(size_t)(side * 2048 + i) * 256 + h * 32 + lane] = out;
}

// residual: out = a*x + b*y (scalars read from device)
__global__ void k_axpby(const float* __restrict__ x, const float* __restrict__ y,
                        const float* __restrict__ a, const float* __restrict__ b,
                        float* __restrict__ out, int n) {
  int i = blockIdx.x * blockDim.x + threadIdx.x;
  if (i < n) out[i] = a[0] * x[i] + b[0] * y[i];
}

// ---------------------------------------------------------------------------
// Host orchestration
// ---------------------------------------------------------------------------
extern "C" void kernel_launch(void* const* d_in, const int* in_sizes, int n_in,
                              void* d_out, int out_size, void* d_ws, size_t ws_size,
                              hipStream_t stream) {
  const float* geo_feat = (const float*)d_in[0];   // [2048,1536]
  const float* sem_feat = (const float*)d_in[1];   // [2048,512]
  const float* rsa_feat = (const float*)d_in[2];   // [2048,64]
  const float* pos      = (const float*)d_in[3];   // [2048,3]
  const float* geo_w = (const float*)d_in[4];   const float* geo_b = (const float*)d_in[5];
  const float* geo_lnw = (const float*)d_in[6]; const float* geo_lnb = (const float*)d_in[7];
  const float* sem_w = (const float*)d_in[8];   const float* sem_b = (const float*)d_in[9];
  const float* sem_lnw = (const float*)d_in[10];const float* sem_lnb = (const float*)d_in[11];
  const float* rsa_w = (const float*)d_in[12];  const float* rsa_b = (const float*)d_in[13];
  const float* rsa_lnw = (const float*)d_in[14];const float* rsa_lnb = (const float*)d_in[15];
  const float* qg_w = (const float*)d_in[16];   const float* qg_b = (const float*)d_in[17];
  const float* qs_w = (const float*)d_in[18];   const float* qs_b = (const float*)d_in[19];
  const float* k_w = (const float*)d_in[20];    const float* k_b = (const float*)d_in[21];
  const float* v_w = (const float*)d_in[22];    const float* v_b = (const float*)d_in[23];
  const float* out_w = (const float*)d_in[24];  const float* out_b = (const float*)d_in[25];
  const float* sw_geo = (const float*)d_in[26]; const float* sw_sem = (const float*)d_in[27];
  const float* alpha_g = (const float*)d_in[28];const float* beta_g = (const float*)d_in[29];
  const float* alpha_s = (const float*)d_in[30];const float* beta_s = (const float*)d_in[31];
  const float* rt_w = (const float*)d_in[32];   const float* rt_b = (const float*)d_in[33];
  const float* rt_lnw = (const float*)d_in[34]; const float* rt_lnb = (const float*)d_in[35];
  const float* f1_w = (const float*)d_in[36];   const float* f1_b = (const float*)d_in[37];
  const float* f1_lnw = (const float*)d_in[38]; const float* f1_lnb = (const float*)d_in[39];
  const float* f2_w = (const float*)d_in[40];   const float* f2_b = (const float*)d_in[41];
  const float* f2_lnw = (const float*)d_in[42]; const float* f2_lnb = (const float*)d_in[43];
  const float* ln1_w = (const float*)d_in[44];  const float* ln1_b = (const float*)d_in[45];
  const float* ln2_w = (const float*)d_in[46];  const float* ln2_b = (const float*)d_in[47];

  (void)in_sizes; (void)n_in; (void)out_size; (void)ws_size;

  // bump allocator over d_ws
  char* wsb = (char*)d_ws;
  size_t off = 0;
  auto alloc = [&](size_t bytes) -> void* {
    void* p = wsb + off;
    off = (off + bytes + 255) & ~(size_t)255;
    return p;
  };
  auto allocB = [&](size_t n) { return (__bf16*)alloc(n * 2); };
  auto allocF = [&](size_t n) { return (float*)alloc(n * 4); };

  // bf16 weights
  __bf16* geoWb = allocB(256 * 1536);
  __bf16* semWb = allocB(256 * 512);
  __bf16* rsaWb = allocB(256 * 64);
  __bf16* qgWb  = allocB(256 * 256);
  __bf16* qsWb  = allocB(256 * 256);
  __bf16* kWb   = allocB(256 * 256);
  __bf16* vWb   = allocB(256 * 256);
  __bf16* outWb = allocB(256 * 256);
  __bf16* rtWb  = allocB(256 * 256);
  __bf16* f1Wb  = allocB(512 * 768);
  __bf16* f2Wb  = allocB(256 * 512);
  // bf16 activations
  __bf16* geoFb = allocB((size_t)2048 * 1536);
  __bf16* semFb = allocB((size_t)2048 * 512);
  __bf16* rsaFb = allocB((size_t)2048 * 64);
  __bf16* jointB = allocB((size_t)4096 * 256);   // gp_bf rows 0..2047, sp_bf rows 2048..4095
  __bf16* rpB   = allocB((size_t)2048 * 256);
  __bf16* QB    = allocB((size_t)2 * 2048 * 256);
  __bf16* KB    = allocB((size_t)4096 * 256);
  __bf16* VT    = allocB((size_t)256 * 4096);
  __bf16* mixB  = allocB((size_t)2 * 2048 * 256);
  __bf16* combB = allocB((size_t)2048 * 768);
  __bf16* hB    = allocB((size_t)2048 * 512);
  // f32 buffers
  float* gp  = allocF((size_t)2048 * 256);
  float* sp  = allocF((size_t)2048 * 256);
  float* rp  = allocF((size_t)2048 * 256);
  float* Kf  = allocF((size_t)4096 * 256);
  float* Vf  = allocF((size_t)4096 * 256);
  float* Qf  = allocF((size_t)2 * 2048 * 256);
  float* pre = allocF((size_t)4096 * 256);       // reused GEMM scratch (>= 2048*512)
  float* Mws = allocF((size_t)2 * 8 * 2048);
  float* Lws = allocF((size_t)2 * 8 * 2048);
  float* Ows = allocF((size_t)2 * 8 * 2048 * 32);
  float* mixF = allocF((size_t)2 * 2048 * 256);
  float* aF   = allocF((size_t)2 * 2048 * 256);
  int* knn = (int*)alloc((size_t)2048 * 24 * 4);

  auto cvt = [&](const float* x, __bf16* y, size_t n) {
    k_f2b<<<dim3((unsigned)((n + 255) / 256)), 256, 0, stream>>>(x, y, (int)n);
  };
  auto gemm = [&](const __bf16* A, const __bf16* W, const float* bias, float* C,
                  int M, int N, int K) {
    k_gemm<<<dim3(N / 64, M / 64), 128, 0, stream>>>(A, W, bias, C, M, N, K);
  };
  auto ln = [&](const float* X, int W, const float* g, const float* b,
                float* Yf, int fs, int fc, __bf16* Yb, int bs, int bc,
                int leaky, int rows) {
    k_ln<<<dim3((rows + 7) / 8), 256, 0, stream>>>(X, W, g, b, Yf, fs, fc, Yb, bs, bc,
                                                   leaky, rows);
  };

  // --- converts ---
  cvt(geo_w, geoWb, 256 * 1536);  cvt(sem_w, semWb, 256 * 512);
  cvt(rsa_w, rsaWb, 256 * 64);    cvt(qg_w, qgWb, 256 * 256);
  cvt(qs_w, qsWb, 256 * 256);     cvt(k_w, kWb, 256 * 256);
  cvt(v_w, vWb, 256 * 256);       cvt(out_w, outWb, 256 * 256);
  cvt(rt_w, rtWb, 256 * 256);     cvt(f1_w, f1Wb, 512 * 768);
  cvt(f2_w, f2Wb, 256 * 512);
  cvt(geo_feat, geoFb, (size_t)2048 * 1536);
  cvt(sem_feat, semFb, (size_t)2048 * 512);
  cvt(rsa_feat, rsaFb, (size_t)2048 * 64);

  // --- input projections + LN (bf16 copies land in jointB / rpB) ---
  gemm(geoFb, geoWb, geo_b, pre, 2048, 256, 1536);
  ln(pre, 256, geo_lnw, geo_lnb, gp, 256, 0, jointB, 256, 0, 0, 2048);
  gemm(semFb, semWb, sem_b, pre, 2048, 256, 512);
  ln(pre, 256, sem_lnw, sem_lnb, sp, 256, 0, jointB + (size_t)2048 * 256, 256, 0, 0, 2048);
  gemm(rsaFb, rsaWb, rsa_b, pre, 2048, 256, 64);
  ln(pre, 256, rsa_lnw, rsa_lnb, rp, 256, 0, rpB, 256, 0, 0, 2048);

  // --- K / V projections + per-head LN ---
  gemm(jointB, kWb, k_b, pre, 4096, 256, 256);
  k_headln<<<dim3(4096), 256, 0, stream>>>(pre, Kf, 32768);
  cvt(Kf, KB, (size_t)4096 * 256);
  gemm(jointB, vWb, v_b, pre, 4096, 256, 256);
  k_headln<<<dim3(4096), 256, 0, stream>>>(pre, Vf, 32768);
  k_vt<<<dim3(4096), 256, 0, stream>>>(Vf, VT);

  // --- queries ---
  gemm(jointB, qgWb, qg_b, Qf, 2048, 256, 256);                      // Qg from gp_bf
  gemm(jointB + (size_t)2048 * 256, qsWb, qs_b, Qf + (size_t)2048 * 256,
       2048, 256, 256);                                              // Qs from sp_bf
  cvt(Qf, QB, (size_t)2 * 2048 * 256);

  // --- KNN, dense flash, corrections+mix ---
  k_knn<<<dim3(2048), 32, 0, stream>>>(pos, knn);
  k_flash<<<dim3(512), 128, 0, stream>>>(QB, KB, VT, Mws, Lws, Ows);
  k_mix<<<dim3(4096), 256, 0, stream>>>(Qf, Kf, Vf, knn, Mws, Lws, Ows,
                                        sw_geo, sw_sem, mixF);
  cvt(mixF, mixB, (size_t)2 * 2048 * 256);

  // --- output projection ---
  gemm(mixB, outWb, out_b, aF, 2048, 256, 256);
  gemm(mixB + (size_t)2048 * 256, outWb, out_b, aF + (size_t)2048 * 256, 2048, 256, 256);

  // --- residual + LN -> comb[:,0:256] and comb[:,256:512] (bf16, stride 768) ---
  k_axpby<<<dim3(2048), 256, 0, stream>>>(gp, aF, alpha_g, beta_g, pre, 2048 * 256);
  ln(pre, 256, ln1_w, ln1_b, nullptr, 0, 0, combB, 768, 0, 0, 2048);
  k_axpby<<<dim3(2048), 256, 0, stream>>>(sp, aF + (size_t)2048 * 256, alpha_s, beta_s,
                                          pre, 2048 * 256);
  ln(pre, 256, ln2_w, ln2_b, nullptr, 0, 0, combB, 768, 256, 0, 2048);

  // --- rsa transform -> comb[:,512:768] (leaky) ---
  gemm(rpB, rtWb, rt_b, pre, 2048, 256, 256);
  ln(pre, 256, rt_lnw, rt_lnb, nullptr, 0, 0, combB, 768, 512, 1, 2048);

  // --- fusion MLP ---
  gemm(combB, f1Wb, f1_b, pre, 2048, 512, 768);
  ln(pre, 512, f1_lnw, f1_lnb, nullptr, 0, 0, hB, 512, 0, 1, 2048);
  gemm(hB, f2Wb, f2_b, pre, 2048, 256, 512);
  ln(pre, 256, f2_lnw, f2_lnb, (float*)d_out, 256, 0, nullptr, 0, 0, 1, 2048);
}
